// SimplifiedTransformerBlock_21096879358040
// MI455X (gfx1250) — matrix operand order repaired and checked by reading, not compiled, not run on hardware
//
// SimplifiedTransformerBlock for MI455X (gfx1250, wave32, WMMA bf16 f32-acc).
//
// Pipeline (all on `stream`):
//  1. transpose+bf16 weights (qk_w, w1, w2)       -> Wt[N,K] bf16
//  2. rmsnorm(x, scale1)                          -> h f32 + h bf16
//  3. transpose h per batch                       -> vT[B,D,S] bf16 (V operand)
//  4. cumsum_s h                                  -> vcum f32 (uniform-causal term)
//  5. WMMA GEMM h@qk_w + qk_b                     -> qk bf16 [8192,1536]
//  6. flash attention (S^T trick) + shaped mix + residual -> x2 f32
//  7. rmsnorm(x2, scale2)                         -> h2 bf16
//  8. WMMA GEMM h2@w1 + b1, LeakyReLU             -> m1 bf16 [8192,3072]
//  9. WMMA GEMM m1@w2 + b2 + x2                   -> d_out f32
//
// GEMM: each wave owns a 32x64 C tile (2 A-frags x 4 B-frags = 8 WMMA per
// 32-wide K step) for ~2.7x the FLOP/byte of a 16x16-per-wave design; block
// is 8 waves arranged 4(M) x 2(N) -> 128x128 block tile.

#include <hip/hip_runtime.h>
#include <hip/hip_bf16.h>

#define D_ 768
#define S_ 2048
#define B_ 4
#define H_ 12
#define M_ (B_ * S_) /* 8192 rows */

typedef __bf16 bf16_t;
typedef __attribute__((ext_vector_type(16))) __bf16 v16bf;
typedef __attribute__((ext_vector_type(8)))  __bf16 v8bf;
typedef __attribute__((ext_vector_type(8)))  float  v8f;

static __device__ __forceinline__ v8f v8f_zero() {
  v8f z = {0.f, 0.f, 0.f, 0.f, 0.f, 0.f, 0.f, 0.f};
  return z;
}

static __device__ __forceinline__ v16bf pack16(v8bf lo, v8bf hi) {
  v16bf r;
#pragma unroll
  for (int e = 0; e < 8; ++e) { r[e] = lo[e]; r[e + 8] = hi[e]; }
  return r;
}

static __device__ __forceinline__ v8f wmma_bf16(v16bf a, v16bf b, v8f c) {
  // D = A(16x32 bf16) * B(32x16 bf16) + C(16x16 f32)
  return __builtin_amdgcn_wmma_f32_16x16x32_bf16(false, a, false, b, (short)0, c,
                                                 false, false);
}

// ---------------------------------------------------------------------------
// Weight transpose + f32->bf16:  Wt[n*K + k] = W[k*N + n]
// ---------------------------------------------------------------------------
__global__ __launch_bounds__(256) void wtrans_kernel(const float* __restrict__ W,
                                                     bf16_t* __restrict__ Wt,
                                                     int K, int N) {
  long long i = (long long)blockIdx.x * 256 + threadIdx.x;
  if (i >= (long long)K * N) return;
  int k = (int)(i % K);
  long long n = i / K;
  Wt[i] = (bf16_t)W[(long long)k * N + n];
}

// ---------------------------------------------------------------------------
// RMSNorm: out = scale * x / (sqrt(mean(x^2)) + eps); writes f32 and bf16.
// One 256-thread block per row of 768.
// ---------------------------------------------------------------------------
__global__ __launch_bounds__(256) void rmsnorm_kernel(const float* __restrict__ x,
                                                      const float* __restrict__ scale,
                                                      float* __restrict__ hf,
                                                      bf16_t* __restrict__ hb) {
  __shared__ float red[8];
  const int row = blockIdx.x;
  const int tid = threadIdx.x;
  const float* xr = x + (size_t)row * D_;
  float v0 = xr[tid], v1 = xr[tid + 256], v2 = xr[tid + 512];
  float ss = v0 * v0 + v1 * v1 + v2 * v2;
#pragma unroll
  for (int m = 16; m > 0; m >>= 1) ss += __shfl_xor(ss, m, 32);
  if ((tid & 31) == 0) red[tid >> 5] = ss;
  __syncthreads();
  float tot = 0.f;
#pragma unroll
  for (int i = 0; i < 8; ++i) tot += red[i];
  float rms = sqrtf(tot * (1.0f / (float)D_));
  float inv = 1.0f / (rms + 1e-8f);
  size_t base = (size_t)row * D_;
#pragma unroll
  for (int j = 0; j < 3; ++j) {
    int d = tid + j * 256;
    float hv = scale[d] * xr[d] * inv;
    hf[base + d] = hv;
    hb[base + d] = (bf16_t)hv;
  }
}

// ---------------------------------------------------------------------------
// vT[b][d][s] = bf16(h[b][s][d])  (V operand, contiguous along keys)
// ---------------------------------------------------------------------------
__global__ __launch_bounds__(256) void vtrans_kernel(const float* __restrict__ hf,
                                                     bf16_t* __restrict__ vT) {
  long long i = (long long)blockIdx.x * 256 + threadIdx.x;
  if (i >= (long long)B_ * D_ * S_) return;
  int s = (int)(i % S_);
  long long t = i / S_;
  int d = (int)(t % D_);
  int b = (int)(t / D_);
  vT[i] = (bf16_t)hf[((size_t)b * S_ + s) * D_ + d];
}

// ---------------------------------------------------------------------------
// vcum[b][s][d] = sum_{s'<=s} h[b][s'][d]   (for the uniform-causal term)
// ---------------------------------------------------------------------------
__global__ __launch_bounds__(256) void cumsum_kernel(const float* __restrict__ hf,
                                                     float* __restrict__ vcum) {
  int b = blockIdx.x / 3;
  int d = (blockIdx.x % 3) * 256 + threadIdx.x;
  const float* p = hf + (size_t)b * S_ * D_ + d;
  float* q = vcum + (size_t)b * S_ * D_ + d;
  float acc = 0.f;
  for (int s = 0; s < S_; ++s) {
    acc += p[(size_t)s * D_];
    q[(size_t)s * D_] = acc;
  }
}

// ---------------------------------------------------------------------------
// WMMA GEMM: C[M,N] = A[M,K](bf16,row-major) * Wt[N,K](bf16) + bias
// 8 waves/block arranged 4(M) x 2(N); each wave computes a 32x64 C tile
// (2 A-frags x 4 B-frags -> 8 WMMA per 32-wide K step). Block tile 128x128.
// mode bit0: LeakyReLU(0.01). Optional f32 residual, f32 and/or bf16 output.
// ---------------------------------------------------------------------------
__global__ __launch_bounds__(256) void gemm_bf16_kernel(
    const bf16_t* __restrict__ A, const bf16_t* __restrict__ Wt,
    const float* __restrict__ bias, const float* __restrict__ resid,
    float* __restrict__ outF, bf16_t* __restrict__ outB,
    int M, int N, int K, int mode) {
  const int tid  = threadIdx.x;
  const int lane = tid & 31;
  const int wid  = tid >> 5;
  const int half = lane >> 4;
  const int l16  = lane & 15;
  const int m0 = blockIdx.y * 128 + (wid & 3) * 32;
  const int n0 = blockIdx.x * 128 + (wid >> 2) * 64;

  const bf16_t* arow[2];
  arow[0] = A + (size_t)(m0 + l16) * K;
  arow[1] = arow[0] + (size_t)16 * K;
  const bf16_t* brow[4];
#pragma unroll
  for (int j = 0; j < 4; ++j) brow[j] = Wt + (size_t)(n0 + j * 16 + l16) * K;

  v8f c[2][4];
#pragma unroll
  for (int i = 0; i < 2; ++i)
#pragma unroll
    for (int j = 0; j < 4; ++j) c[i][j] = v8f_zero();

  for (int k0 = 0; k0 < K; k0 += 32) {
    v16bf a[2], b[4];
#pragma unroll
    for (int i = 0; i < 2; ++i) {
      // A frag: elems 0..7 -> k = k0+8*half+e ; 8..15 -> k = k0+16+8*half+e
      a[i] = pack16(*(const v8bf*)(arow[i] + k0 + 8 * half),
                    *(const v8bf*)(arow[i] + k0 + 16 + 8 * half));
    }
#pragma unroll
    for (int j = 0; j < 4; ++j) {
      // B frag: elems 0..7 -> k = k0+8*half+e ; 8..15 -> k = k0+16+8*half+e (column n = l16 of W)
      b[j] = pack16(*(const v8bf*)(brow[j] + k0 + 8 * half),
                    *(const v8bf*)(brow[j] + k0 + 16 + 8 * half));
    }
#pragma unroll
    for (int i = 0; i < 2; ++i)
#pragma unroll
      for (int j = 0; j < 4; ++j) c[i][j] = wmma_bf16(a[i], b[j], c[i][j]);
  }

#pragma unroll
  for (int j = 0; j < 4; ++j) {
    const int ncol = n0 + j * 16 + l16;
    const float bv = bias ? bias[ncol] : 0.f;
#pragma unroll
    for (int i = 0; i < 2; ++i) {
#pragma unroll
      for (int r = 0; r < 8; ++r) {
        int mrow = m0 + i * 16 + r + 8 * half; // C: m=r+8*(lane/16), n=lane%16
        float v = c[i][j][r] + bv;
        if (mode & 1) v = (v > 0.f) ? v : 0.01f * v;
        size_t idx = (size_t)mrow * N + ncol;
        if (resid) v += resid[idx];
        if (outF) outF[idx] = v;
        if (outB) outB[idx] = (bf16_t)v;
      }
    }
  }
}

// ---------------------------------------------------------------------------
// Shaped flash attention. One wave per (b, head, 16-query tile).
// Computes S^T = K_tile * Q^T so the exp'd score registers are already in
// the A-fragment layout needed by the P*V WMMA (no cross-lane movement).
// Epilogue: x2 = x + g_resid*softmaxV + g_skip*h - g_centre*cumsum(h)/(q+1)
// ---------------------------------------------------------------------------
__global__ __launch_bounds__(256) void attn_kernel(
    const bf16_t* __restrict__ qkb,  // [B*S,1536]; q at col h*64, k at 768+h*64
    const bf16_t* __restrict__ vT,   // [B,768,S]
    const float* __restrict__ hf,    // [B,S,768]
    const float* __restrict__ vcum,  // [B,S,768]
    const float* __restrict__ x,     // [B,S,768]
    const float* __restrict__ g_res, const float* __restrict__ g_skip,
    const float* __restrict__ g_cen, float* __restrict__ x2) {
  const int wid_g = blockIdx.x * 8 + (threadIdx.x >> 5);
  if (wid_g >= B_ * H_ * (S_ / 16)) return; // wave-uniform

  const int lane = threadIdx.x & 31;
  const int half = lane >> 4;
  const int l16  = lane & 15;
  const int qt = wid_g & 127;
  const int hh = (wid_g >> 7) % H_;
  const int b  = wid_g / (128 * H_);
  const int q0 = qt * 16;
  const size_t RS = 2 * D_; // 1536

  const bf16_t* qbase = qkb + (size_t)b * S_ * RS + (size_t)hh * 64;
  const bf16_t* kbase = qbase + D_;

  // Q^T B-fragments (one per 32-wide k-step of head dim), loop-invariant.
  v16bf qf[2];
#pragma unroll
  for (int ks = 0; ks < 2; ++ks) {
    const bf16_t* p = qbase + (size_t)(q0 + l16) * RS + ks * 32 + 8 * half;
    qf[ks] = pack16(*(const v8bf*)p, *(const v8bf*)(p + 16));
  }

  float mrun = -1e30f, lrun = 0.f;
  v8f acc[4];
#pragma unroll
  for (int t = 0; t < 4; ++t) acc[t] = v8f_zero();

  for (int k0 = 0; k0 <= q0 + 15; k0 += 32) {
    // scores^T: two 16-key tiles
    v8f sc[2];
#pragma unroll
    for (int t = 0; t < 2; ++t) {
      v8f c = v8f_zero();
#pragma unroll
      for (int ks = 0; ks < 2; ++ks) {
        const bf16_t* p =
            kbase + (size_t)(k0 + t * 16 + l16) * RS + ks * 32 + 8 * half;
        c = wmma_bf16(pack16(*(const v8bf*)p, *(const v8bf*)(p + 16)), qf[ks], c);
      }
      sc[t] = c;
    }
    // causal mask + tile max (this lane owns query q0+l16; key = k0+16t+r+8*half)
    float tmax = -1e30f;
#pragma unroll
    for (int t = 0; t < 2; ++t)
#pragma unroll
      for (int r = 0; r < 8; ++r) {
        int kg = k0 + t * 16 + r + 8 * half;
        float s = (kg <= q0 + l16) ? sc[t][r] : -1e30f;
        sc[t][r] = s;
        tmax = fmaxf(tmax, s);
      }
    tmax = fmaxf(tmax, __shfl_xor(tmax, 16, 32));
    float mnew  = fmaxf(mrun, tmax);
    float alpha = __expf(mrun - mnew);
    // exp + P fragment (already A-layout for the P*V WMMA)
    v16bf pA;
    float lsum = 0.f;
#pragma unroll
    for (int t = 0; t < 2; ++t)
#pragma unroll
      for (int r = 0; r < 8; ++r) {
        float pv = __expf(sc[t][r] - mnew);
        lsum += pv;
        pA[t * 8 + r] = (bf16_t)pv;
      }
    lsum += __shfl_xor(lsum, 16, 32);
    lrun = lrun * alpha + lsum;
    mrun = mnew;
    // rescale output accumulators (query of acc row r is r+8*half)
#pragma unroll
    for (int r = 0; r < 8; ++r) {
      float ar = __shfl(alpha, r + 8 * half, 32);
      acc[0][r] *= ar; acc[1][r] *= ar; acc[2][r] *= ar; acc[3][r] *= ar;
    }
    // O += P(16x32) * V(32x64): 4 dim-tiles
#pragma unroll
    for (int t = 0; t < 4; ++t) {
      const bf16_t* p = vT + ((size_t)b * D_ + hh * 64 + t * 16 + l16) * S_ +
                        k0 + 8 * half;
      acc[t] = wmma_bf16(pA, pack16(*(const v8bf*)p, *(const v8bf*)(p + 16)), acc[t]);
    }
  }

  const float linv = 1.0f / lrun;
  const float gr = g_res[hh], gs = g_skip[hh], gc = g_cen[hh];
#pragma unroll
  for (int r = 0; r < 8; ++r) {
    int qr = r + 8 * half;
    float li = __shfl(linv, qr, 32);
    int qg = q0 + qr;
    float invq = 1.0f / (float)(qg + 1);
    size_t base = ((size_t)b * S_ + qg) * D_ + hh * 64;
#pragma unroll
    for (int t = 0; t < 4; ++t) {
      size_t idx = base + t * 16 + l16;
      float o = gr * acc[t][r] * li + gs * hf[idx] - gc * vcum[idx] * invq;
      x2[idx] = x[idx] + o;
    }
  }
}

// ---------------------------------------------------------------------------
extern "C" void kernel_launch(void* const* d_in, const int* in_sizes, int n_in,
                              void* d_out, int out_size, void* d_ws, size_t ws_size,
                              hipStream_t stream) {
  (void)in_sizes; (void)n_in; (void)out_size; (void)ws_size;
  const float* x      = (const float*)d_in[0];
  const float* scale1 = (const float*)d_in[1];
  const float* qk_w   = (const float*)d_in[2];
  const float* qk_b   = (const float*)d_in[3];
  const float* g_res  = (const float*)d_in[4];
  const float* g_skp  = (const float*)d_in[5];
  const float* g_cen  = (const float*)d_in[6];
  const float* scale2 = (const float*)d_in[7];
  const float* w1     = (const float*)d_in[8];
  const float* b1     = (const float*)d_in[9];
  const float* w2     = (const float*)d_in[10];
  const float* b2     = (const float*)d_in[11];
  float* out = (float*)d_out;

  char* ws = (char*)d_ws;
  float*  hf   = (float*)(ws + 0);            // 25165824 B
  bf16_t* hb   = (bf16_t*)(ws + 25165824);    // 12582912 B
  bf16_t* qkwT = (bf16_t*)(ws + 37748736);    //  2359296 B
  bf16_t* w1T  = (bf16_t*)(ws + 40108032);    //  4718592 B
  bf16_t* w2T  = (bf16_t*)(ws + 44826624);    //  4718592 B
  bf16_t* qkb  = (bf16_t*)(ws + 49545216);    // 25165824 B
  bf16_t* vT   = (bf16_t*)(ws + 74711040);    // 12582912 B
  float*  vcum = (float*)(ws + 87293952);     // 25165824 B
  float*  x2   = (float*)(ws + 112459776);    // 25165824 B (peak ~138 MB)
  bf16_t* m1   = (bf16_t*)(ws + 49545216);    // 50331648 B, aliases qkb/vT/vcum
                                              // (all dead after attn_kernel)
  dim3 blk(256);

  // 1. stage weights transposed bf16
  wtrans_kernel<<<dim3((D_ * 2 * D_ + 255) / 256), blk, 0, stream>>>(qk_w, qkwT, D_, 2 * D_);
  wtrans_kernel<<<dim3((D_ * 4 * D_ + 255) / 256), blk, 0, stream>>>(w1, w1T, D_, 4 * D_);
  wtrans_kernel<<<dim3((4 * D_ * D_ + 255) / 256), blk, 0, stream>>>(w2, w2T, 4 * D_, D_);
  // 2. rmsnorm 1
  rmsnorm_kernel<<<dim3(M_), blk, 0, stream>>>(x, scale1, hf, hb);
  // 3-4. attention operand prep
  vtrans_kernel<<<dim3((B_ * D_ * S_ + 255) / 256), blk, 0, stream>>>(hf, vT);
  cumsum_kernel<<<dim3(B_ * 3), blk, 0, stream>>>(hf, vcum);
  // 5. fused QK projection -> bf16
  gemm_bf16_kernel<<<dim3((2 * D_) / 128, M_ / 128), blk, 0, stream>>>(
      hb, qkwT, qk_b, nullptr, nullptr, qkb, M_, 2 * D_, D_, 0);
  // 6. shaped flash attention + residual -> x2
  attn_kernel<<<dim3(B_ * H_ * (S_ / 16) / 8), blk, 0, stream>>>(
      qkb, vT, hf, vcum, x, g_res, g_skp, g_cen, x2);
  // 7. rmsnorm 2 (reuses hf/hb)
  rmsnorm_kernel<<<dim3(M_), blk, 0, stream>>>(x2, scale2, hf, hb);
  // 8. MLP up + LeakyReLU -> bf16
  gemm_bf16_kernel<<<dim3((4 * D_) / 128, M_ / 128), blk, 0, stream>>>(
      hb, w1T, b1, nullptr, nullptr, m1, M_, 4 * D_, D_, 1);
  // 9. MLP down + bias + residual -> d_out (f32)
  gemm_bf16_kernel<<<dim3(D_ / 128, M_ / 128), blk, 0, stream>>>(
      m1, w2T, b2, x2, out, nullptr, M_, D_, 4 * D_, 0);
}
